// Net_83691732730524
// MI455X (gfx1250) — compile-verified, matrix operand order checked by reading, not run on hardware
//
#include <hip/hip_runtime.h>
#include <math.h>
#include <stdint.h>

// ---------------------------------------------------------------------------
// Recurrent MLP (77->500->500->500->500->19, SELU x4 + tanh), 10 timesteps.
// One kernel launch per timestep (recurrence round-trips through d_out, so
// no loop-invariant weight loads can be hoisted/spilled). Each workgroup
// (512 threads = 16 wave32) owns 32 batch rows; activations ping-pong
// between two 32x520 bf16 LDS buffers (AS3 -> ds_load_b128). Weights bf16,
// padded, L2-resident, streamed as WMMA B-fragments (AS1 -> global_load_b128)
// with an explicit two-stage ping-pong pipeline (no rotation movs).
// Matrix core: v_wmma_f32_16x16x32_bf16. Branchless SELU/tanh epilogues.
// ---------------------------------------------------------------------------

typedef __attribute__((ext_vector_type(16))) __bf16 v16bf;
typedef __attribute__((ext_vector_type(8)))  __bf16 v8bf;
typedef __attribute__((ext_vector_type(8)))  float  v8f;

#define ASG __attribute__((address_space(1)))
#define ASL __attribute__((address_space(3)))
typedef ASG __bf16 gbf16;
typedef ASL __bf16 lbf16;
typedef ASG float  gfloat;
typedef ASL v8bf   lv8bf_t;
typedef ASG v8bf   gv8bf;
typedef ASL v8bf   lv8bf;

#define MT        32      // batch rows per workgroup
#define LDA       520     // LDS row stride in halves (16B aligned, bank-spread)
#define NTHREADS  512     // 16 waves of 32

__constant__ float cMEAN[58] = {
  0.5533f,0.5563f,0.5872f,0.5788f,0.5771f,0.5916f,0.5669f,0.5585f,0.5965f,0.5861f,
  0.6195f,0.6159f,0.5876f,0.6443f,0.5022f,0.5607f,0.5544f,0.7234f,0.6282f,0.2103f,
  0.2046f,0.2499f,0.2619f,0.2328f,0.2498f,0.2324f,0.2011f,0.2571f,0.2555f,0.2775f,
  0.2529f,0.2004f,0.3126f,0.1285f,0.1803f,0.1553f,0.3783f,0.2942f,0.0928f,0.0841f,
  0.0964f,0.1013f,0.0788f,0.0556f,0.0303f,0.0141f,0.0228f,0.0410f,0.0423f,0.0336f,
  0.0048f,0.0041f,-0.0105f,-0.0060f,-0.0052f,0.0145f,0.0092f,0.5f
};
__constant__ float cSTD[58] = {
  0.4972f,0.4968f,0.4924f,0.4938f,0.4940f,0.4916f,0.4955f,0.4966f,0.4906f,0.4926f,
  0.4855f,0.4864f,0.4923f,0.4788f,0.5000f,0.4963f,0.4971f,0.4473f,0.4833f,0.9618f,
  0.9478f,0.9080f,0.8957f,0.8811f,0.8559f,0.8667f,0.8335f,0.8469f,0.8096f,0.8405f,
  0.7865f,0.7515f,0.7853f,0.8256f,0.7923f,0.7239f,0.6611f,0.6207f,0.7900f,0.7303f,
  0.6827f,0.6167f,0.5790f,0.5346f,0.4577f,0.4234f,0.3888f,0.3376f,0.3059f,0.2776f,
  0.2552f,0.1699f,0.1398f,0.1294f,0.1117f,0.0615f,0.0539f,0.3192f
};

// 16x32 bf16 operand fragment. ISA 16-bit layout: lanes 0-15 hold K=0..7
// (v0..3) and K=16..23 (v4..7); lanes 16-31 hold K=8..15 and K=24..31.
// Caller passes &M[row_of_this_lane][kbase + (lane>>4)*8].
__device__ __forceinline__ v16bf load_fragL(const lbf16* p) {
  const v8bf lo = *(const lv8bf*)(p);
  const v8bf hi = *(const lv8bf*)(p + 16);
  return __builtin_shufflevector(lo, hi, 0,1,2,3,4,5,6,7,8,9,10,11,12,13,14,15);
}
__device__ __forceinline__ v16bf load_fragG(const gbf16* p) {
  const v8bf lo = *(const gv8bf*)(p);
  const v8bf hi = *(const gv8bf*)(p + 16);
  return __builtin_shufflevector(lo, hi, 0,1,2,3,4,5,6,7,8,9,10,11,12,13,14,15);
}

__device__ __forceinline__ v8f wmma_bf16(v16bf a, v16bf b, v8f c) {
  return __builtin_amdgcn_wmma_f32_16x16x32_bf16(
      false, a, false, b, (short)0, c, false, false);
}

// Branchless SELU: exp computed unconditionally on clamped arg, then select.
__device__ __forceinline__ float selu_f(float x) {
  const float e   = __expf(fminf(x, 0.f));
  const float neg = 1.7580993408473766f * (e - 1.f);   // scale*alpha*(e^x - 1)
  const float pos = 1.0507009873554805f * x;
  return x > 0.f ? pos : neg;
}
// Branchless tanh: 1 - 2/(e^{2x}+1); saturates correctly via exp over/underflow.
__device__ __forceinline__ float tanh_f(float x) {
  const float e = __expf(2.f * x);
  return 1.f - 2.f * __builtin_amdgcn_rcpf(e + 1.f);
}

#define LDB4(dst, kk)                                                      \
  dst##0 = load_fragG(W + bofs + 0u * 16u * (unsigned)K + (unsigned)(kk)); \
  dst##1 = load_fragG(W + bofs + 1u * 16u * (unsigned)K + (unsigned)(kk)); \
  dst##2 = load_fragG(W + bofs + 2u * 16u * (unsigned)K + (unsigned)(kk)); \
  dst##3 = load_fragG(W + bofs + 3u * 16u * (unsigned)K + (unsigned)(kk));
#define WMMA4(aa, bb)                                                      \
  acc0 = wmma_bf16(aa, bb##0, acc0);                                       \
  acc1 = wmma_bf16(aa, bb##1, acc1);                                       \
  acc2 = wmma_bf16(aa, bb##2, acc2);                                       \
  acc3 = wmma_bf16(aa, bb##3, acc3);

// sOut[32 x 512] = selu(sIn[32 x K] @ W^T + bias).  W: bf16 (512 x K) padded.
// Wave w: M-tile = (w&1)*16, N-range = (w>>1)*64 (4 tiles of 16).
// Two-stage ping-pong pipeline: sets 0/1 alternate, no rotation copies.
template<int K>
__device__ __forceinline__ void dense_selu(const lbf16* __restrict__ sIn,
                                           lbf16* __restrict__ sOut,
                                           const gbf16* __restrict__ W,
                                           const gfloat* __restrict__ bias,
                                           int wid, int lane) {
  const int mbase = (wid & 1) * 16;
  const int nbase = (wid >> 1) * 64;
  const int l15   = lane & 15;
  const int kh    = (lane >> 4) * 8;
  v8f acc0 = (v8f){0.f,0.f,0.f,0.f,0.f,0.f,0.f,0.f};
  v8f acc1 = acc0, acc2 = acc0, acc3 = acc0;

  const lbf16*   aRow = sIn + (mbase + l15) * LDA + kh;
  const unsigned bofs = (unsigned)(nbase + l15) * (unsigned)K + (unsigned)kh;

  v16bf a0, a1, b00, b01, b02, b03, b10, b11, b12, b13;
  a0 = load_fragL(aRow);
  LDB4(b0, 0)

#pragma unroll 1
  for (int kc = 0; kc + 64 < K; kc += 64) {
    a1 = load_fragL(aRow + kc + 32);
    LDB4(b1, kc + 32)
    WMMA4(a0, b0)
    a0 = load_fragL(aRow + kc + 64);
    LDB4(b0, kc + 64)
    WMMA4(a1, b1)
  }
  a1 = load_fragL(aRow + (K - 32));
  LDB4(b1, K - 32)
  WMMA4(a0, b0)
  WMMA4(a1, b1)

  // C/D layout: VGPR r -> M = r (lanes 0-15) / M = 8+r (lanes 16-31)
  const int mhi = (lane >> 4) * 8;
  v8f accs[4] = {acc0, acc1, acc2, acc3};
#pragma unroll
  for (int j = 0; j < 4; ++j) {
    const int ncol = nbase + j * 16 + l15;
    const float bb = bias[ncol];
#pragma unroll
    for (int r = 0; r < 8; ++r) {
      const int mrow = mbase + r + mhi;
      sOut[mrow * LDA + ncol] = (__bf16)selu_f(accs[j][r] + bb);
    }
  }
}

// Final layer: tanh(sIn[32x512] @ W5^T + b5), N padded 19->32. Waves 0..3.
__device__ __forceinline__ void dense_tanh_out(const lbf16* __restrict__ sIn,
                                               const gbf16* __restrict__ W,
                                               const gfloat* __restrict__ b5,
                                               gfloat* __restrict__ out,
                                               int wid, int lane, int row0, int t) {
  if (wid < 4) {
    const int mbase = (wid & 1) * 16;
    const int nbase = (wid >> 1) * 16;
    const int l15   = lane & 15;
    const int kh    = (lane >> 4) * 8;
    v8f acc = (v8f){0.f,0.f,0.f,0.f,0.f,0.f,0.f,0.f};
    const lbf16*   aRow = sIn + (mbase + l15) * LDA + kh;
    const unsigned bofs = (unsigned)(nbase + l15) * 512u + (unsigned)kh;

    v16bf a0 = load_fragL(aRow);
    v16bf b0 = load_fragG(W + bofs);
    v16bf a1, b1;
#pragma unroll 1
    for (int kc = 0; kc + 64 < 512; kc += 64) {
      a1 = load_fragL(aRow + kc + 32);
      b1 = load_fragG(W + bofs + (unsigned)(kc + 32));
      acc = wmma_bf16(a0, b0, acc);
      a0 = load_fragL(aRow + kc + 64);
      b0 = load_fragG(W + bofs + (unsigned)(kc + 64));
      acc = wmma_bf16(a1, b1, acc);
    }
    a1 = load_fragL(aRow + 480);
    b1 = load_fragG(W + bofs + 480u);
    acc = wmma_bf16(a0, b0, acc);
    acc = wmma_bf16(a1, b1, acc);

    const int mhi  = (lane >> 4) * 8;
    const int ncol = nbase + l15;
    const float bb = b5[ncol];
#pragma unroll
    for (int r = 0; r < 8; ++r) {
      const int mrow = mbase + r + mhi;
      const float v = tanh_f(acc[r] + bb);
      if (ncol < 19) {
        out[(size_t)(row0 + mrow) * 190 + ncol * 10 + t] = v;
      }
    }
  }
}

// One timestep: assemble input (x slice t + prev from out[:, :, t-1]),
// run the 5-layer MLP, write out[:, :, t].
extern "C" __global__ __launch_bounds__(NTHREADS, 1)
void mlp_step(const float* __restrict__ x_,
              const __bf16* __restrict__ Wp1_, const __bf16* __restrict__ Wp2_,
              const __bf16* __restrict__ Wp3_, const __bf16* __restrict__ Wp4_,
              const __bf16* __restrict__ Wp5_,
              const float* __restrict__ bp1_, const float* __restrict__ bp2_,
              const float* __restrict__ bp3_, const float* __restrict__ bp4_,
              const float* __restrict__ bp5_,
              float* __restrict__ out_, int t) {
  const gfloat* x   = (const gfloat*)(uintptr_t)x_;
  const gbf16*  Wp1 = (const gbf16*)(uintptr_t)Wp1_;
  const gbf16*  Wp2 = (const gbf16*)(uintptr_t)Wp2_;
  const gbf16*  Wp3 = (const gbf16*)(uintptr_t)Wp3_;
  const gbf16*  Wp4 = (const gbf16*)(uintptr_t)Wp4_;
  const gbf16*  Wp5 = (const gbf16*)(uintptr_t)Wp5_;
  const gfloat* bp1 = (const gfloat*)(uintptr_t)bp1_;
  const gfloat* bp2 = (const gfloat*)(uintptr_t)bp2_;
  const gfloat* bp3 = (const gfloat*)(uintptr_t)bp3_;
  const gfloat* bp4 = (const gfloat*)(uintptr_t)bp4_;
  const gfloat* bp5 = (const gfloat*)(uintptr_t)bp5_;
  gfloat*       out = (gfloat*)(uintptr_t)out_;

  extern __shared__ char smem[];
  const unsigned sbase = (unsigned)(uintptr_t)smem;  // LDS offset (low 32 bits)
  lbf16* sA = (lbf16*)sbase;                         // 32 x 520 bf16
  lbf16* sB = (lbf16*)(sbase + MT * LDA * 2);        // 32 x 520 bf16

  const int tid  = threadIdx.x;
  const int wid  = tid >> 5;
  const int lane = tid & 31;
  const int row0 = blockIdx.x * MT;

  // Layer-1 input tile: [norm x(:, :, t) | prev | zero-pad] -> 32x128.
  // prev = out[:, :, t-1] for t>0, else normalized x[:, 0:19, 0].
  for (int idx = tid; idx < MT * 128; idx += NTHREADS) {
    const int m = idx >> 7, c = idx & 127;
    const size_t row = (size_t)(row0 + m);
    float v = 0.f;
    if (c < 58) {
      v = (x[row * 580 + c * 10 + t] - cMEAN[c]) / cSTD[c];
    } else if (c < 77) {
      const int n = c - 58;
      if (t == 0) v = (x[row * 580 + n * 10] - cMEAN[n]) / cSTD[n];
      else        v = out[row * 190 + n * 10 + (t - 1)];
    }
    sA[m * LDA + c] = (__bf16)v;
  }
  __syncthreads();

  dense_selu<128>(sA, sB, Wp1, bp1, wid, lane); __syncthreads();
  dense_selu<512>(sB, sA, Wp2, bp2, wid, lane); __syncthreads();
  dense_selu<512>(sA, sB, Wp3, bp3, wid, lane); __syncthreads();
  dense_selu<512>(sB, sA, Wp4, bp4, wid, lane); __syncthreads();
  dense_tanh_out(sA, Wp5, bp5, out, wid, lane, row0, t);
}

// --- weight/bias prep: f32 -> bf16 with zero padding -------------------------
extern "C" __global__ void pad_w(const float* __restrict__ W, __bf16* __restrict__ Wp,
                                 int N0, int K0, int Np, int Kp) {
  for (int i = blockIdx.x * blockDim.x + threadIdx.x; i < Np * Kp;
       i += gridDim.x * blockDim.x) {
    const int n = i / Kp, k = i - n * Kp;
    const float v = (n < N0 && k < K0) ? W[(size_t)n * K0 + k] : 0.f;
    Wp[i] = (__bf16)v;
  }
}

extern "C" __global__ void pad_b(const float* __restrict__ b, float* __restrict__ bp,
                                 int N0, int Np) {
  for (int i = blockIdx.x * blockDim.x + threadIdx.x; i < Np;
       i += gridDim.x * blockDim.x) {
    bp[i] = (i < N0) ? b[i] : 0.f;
  }
}

extern "C" void kernel_launch(void* const* d_in, const int* in_sizes, int n_in,
                              void* d_out, int out_size, void* d_ws, size_t ws_size,
                              hipStream_t stream) {
  (void)in_sizes; (void)n_in; (void)out_size; (void)ws_size;
  const float* x  = (const float*)d_in[0];
  const float* W1 = (const float*)d_in[1];
  const float* b1 = (const float*)d_in[2];
  const float* W2 = (const float*)d_in[3];
  const float* b2 = (const float*)d_in[4];
  const float* W3 = (const float*)d_in[5];
  const float* b3 = (const float*)d_in[6];
  const float* W4 = (const float*)d_in[7];
  const float* b4 = (const float*)d_in[8];
  const float* W5 = (const float*)d_in[9];
  const float* b5 = (const float*)d_in[10];

  char* w = (char*)d_ws;
  __bf16* Wp1 = (__bf16*)w; w += (size_t)512 * 128 * 2;
  __bf16* Wp2 = (__bf16*)w; w += (size_t)512 * 512 * 2;
  __bf16* Wp3 = (__bf16*)w; w += (size_t)512 * 512 * 2;
  __bf16* Wp4 = (__bf16*)w; w += (size_t)512 * 512 * 2;
  __bf16* Wp5 = (__bf16*)w; w += (size_t)32  * 512 * 2;
  float* bp1 = (float*)w; w += 512 * 4;
  float* bp2 = (float*)w; w += 512 * 4;
  float* bp3 = (float*)w; w += 512 * 4;
  float* bp4 = (float*)w; w += 512 * 4;
  float* bp5 = (float*)w; w += 32  * 4;

  pad_w<<<64,  256, 0, stream>>>(W1, Wp1, 500,  77, 512, 128);
  pad_w<<<256, 256, 0, stream>>>(W2, Wp2, 500, 500, 512, 512);
  pad_w<<<256, 256, 0, stream>>>(W3, Wp3, 500, 500, 512, 512);
  pad_w<<<256, 256, 0, stream>>>(W4, Wp4, 500, 500, 512, 512);
  pad_w<<<16,  256, 0, stream>>>(W5, Wp5,  19, 500,  32, 512);
  pad_b<<<2, 256, 0, stream>>>(b1, bp1, 500, 512);
  pad_b<<<2, 256, 0, stream>>>(b2, bp2, 500, 512);
  pad_b<<<2, 256, 0, stream>>>(b3, bp3, 500, 512);
  pad_b<<<2, 256, 0, stream>>>(b4, bp4, 500, 512);
  pad_b<<<1, 32,  0, stream>>>(b5, bp5,  19,  32);

  const size_t ldsBytes = (size_t)(2 * MT * LDA * 2); // 66,560 B
  for (int t = 0; t < 10; ++t) {
    mlp_step<<<65536 / MT, NTHREADS, ldsBytes, stream>>>(
        x, Wp1, Wp2, Wp3, Wp4, Wp5, bp1, bp2, bp3, bp4, bp5, (float*)d_out, t);
  }
}
